// TFGroupViTTokenAssign_59571196395719
// MI455X (gfx1250) — matrix operand order repaired and checked by reading, not compiled, not run on hardware
//
#include <hip/hip_runtime.h>
#include <hip/hip_bf16.h>

// ---------------- problem constants ----------------
#define BB   256
#define LL   196
#define CC   384
#define HH   6
#define GG   64
#define GGO  64
#define TOKD 192
#define FFD  1536
#define CHD  1536
#define LNEPS 1e-5f

// epilogue modes
#define EPI_NONE  0
#define EPI_GELU  1
#define EPI_RES   2
#define EPI_SCALE 3

#if __has_builtin(__builtin_amdgcn_tensor_load_to_lds)
#define HAVE_TDM 1
#else
#define HAVE_TDM 0
#endif

typedef __bf16 bf16x16 __attribute__((ext_vector_type(16)));
typedef float  f32x8   __attribute__((ext_vector_type(8)));
typedef unsigned int u32x4 __attribute__((ext_vector_type(4)));
typedef int i32x4 __attribute__((ext_vector_type(4)));
typedef int i32x8 __attribute__((ext_vector_type(8)));

union FragU { bf16x16 v; uint4 q[2]; };

__device__ __forceinline__ unsigned short f2bf(float f) {
  unsigned int u = __float_as_uint(f);
  unsigned int r = u + 0x7FFFu + ((u >> 16) & 1u);
  return (unsigned short)(r >> 16);
}

__device__ __forceinline__ unsigned int pack2bf(float a, float b) {
  return (unsigned int)f2bf(a) | ((unsigned int)f2bf(b) << 16);
}

__device__ __forceinline__ float gelu_exact(float x) {
  return 0.5f * x * (1.0f + erff(x * 0.70710678118654752440f));
}

__device__ __forceinline__ unsigned lds_off32(const void* p) {
  // generic LDS address: aperture in high 32 bits, byte offset in low 32 bits
  return (unsigned)(unsigned long long)p;
}

// Issue one TDM 2-D tile load: bf16 tile 64x64 from (N,K)-major weight into
// LDS with +16B padding after each 128B row (LDS row stride = 72 elements).
__device__ __forceinline__ void tdm_load_b_tile(const unsigned short* gsrc,
                                                unsigned ldsoff,
                                                int remK, int remN, int ldbK) {
#if HAVE_TDM
  const unsigned long long ga = (unsigned long long)(const void*)gsrc;
  u32x4 g0;
  g0[0] = 1u;                                   // count=1, user mode, no gather
  g0[1] = ldsoff;                               // lds_addr (bytes)
  g0[2] = (unsigned)(ga & 0xFFFFFFFFull);       // global_addr[31:0]
  g0[3] = (unsigned)((ga >> 32) & 0x01FFFFFFull) | (2u << 30);  // addr[56:32], type=2
  i32x8 g1;
  // mask=0 | data_size=1(2B)<<16 | pad_enable<<20 | pad_interval=4(32DW)<<22 | pad_amount=3(4DW)<<25
  g1[0] = (1 << 16) | (1 << 20) | (4 << 22) | (3 << 25);
  const unsigned td0 = (unsigned)remK, td1 = (unsigned)remN;
  g1[1] = (int)((td0 & 0xFFFFu) << 16);                           // tensor_dim0[15:0] @ [63:48]
  g1[2] = (int)(((td0 >> 16) & 0xFFFFu) | ((td1 & 0xFFFFu) << 16)); // dim0 hi | dim1 lo
  g1[3] = (int)(((td1 >> 16) & 0xFFFFu) | (64u << 16));           // dim1 hi | tile_dim0=64
  g1[4] = (int)64;                                                // tile_dim1=64, tile_dim2=0
  g1[5] = (int)(unsigned)ldbK;                                    // tensor_dim0_stride[31:0]
  g1[6] = 0;                                                      // stride hi, dim1_stride lo
  g1[7] = 0;
  const i32x4 gz = {0, 0, 0, 0};
#if defined(__clang_major__) && __clang_major__ >= 23
  const i32x8 gz8 = {0, 0, 0, 0, 0, 0, 0, 0};
  __builtin_amdgcn_tensor_load_to_lds(g0, g1, gz, gz, gz8, 0);
#else
  __builtin_amdgcn_tensor_load_to_lds(g0, g1, gz, gz, 0);
#endif
#else
  (void)gsrc; (void)ldsoff; (void)remK; (void)remN; (void)ldbK;
#endif
}

// ---------------- LayerNorm over last dim (C) ----------------
__global__ __launch_bounds__(256) void ln_kernel(const float* __restrict__ in,
                                                 const float* __restrict__ g,
                                                 const float* __restrict__ b,
                                                 float* __restrict__ out, int C) {
  __shared__ float s1[256], s2[256];
  const int row = blockIdx.x;
  const float* x = in + (size_t)row * C;
  float s = 0.f, ss = 0.f;
  for (int i = threadIdx.x; i < C; i += 256) { float v = x[i]; s += v; ss += v * v; }
  s1[threadIdx.x] = s; s2[threadIdx.x] = ss;
  __syncthreads();
  for (int st = 128; st > 0; st >>= 1) {
    if (threadIdx.x < st) { s1[threadIdx.x] += s1[threadIdx.x + st];
                            s2[threadIdx.x] += s2[threadIdx.x + st]; }
    __syncthreads();
  }
  const float mean = s1[0] / C;
  const float var  = s2[0] / C - mean * mean;
  const float rstd = rsqrtf(var + LNEPS);
  float* o = out + (size_t)row * C;
  for (int i = threadIdx.x; i < C; i += 256)
    o[i] = (x[i] - mean) * rstd * g[i] + b[i];
}

// ------- weight prep: (K,N) f32 -> (N,K) bf16 -------
__global__ __launch_bounds__(256) void wprep_kernel(const float* __restrict__ in,
                                                    unsigned short* __restrict__ out,
                                                    int K, int N) {
  const int i = blockIdx.x * 256 + threadIdx.x;
  if (i >= N * K) return;
  const int n = i / K, k = i % K;
  out[i] = f2bf(in[(size_t)k * N + n]);
}

// ---------------- weight-B GEMM: bf16 (N,K) weights via TDM ----------------
// C[b] = epilogue( A[b](MxK) * W^T + bias ); requires M,N,K multiples of 64.
template <int AT>
__global__ __launch_bounds__(256) void gemm_wb_kernel(
    const float* __restrict__ A, int lda, long long sA,
    const unsigned short* __restrict__ Wb, int ldbK,
    const float* __restrict__ bias, const float* __restrict__ res,
    float* __restrict__ Cm, int ldc, long long sC, int cT,
    int M, int N, int K, int epi) {
  __shared__ __align__(16) unsigned short As[64 * 72];
  __shared__ __align__(16) unsigned short Bs[64 * 72];
  const int tid = threadIdx.x;
  const int wv = tid >> 5, ln = tid & 31;
  const int lm = ln & 15, hi = ln >> 4;
  const int tr = wv & 3;
  const int tc = (wv >> 2) << 1;
  const int m0 = blockIdx.y * 64, n0 = blockIdx.x * 64;
  const float* Ab = A + (size_t)blockIdx.z * sA;

  f32x8 acc0 = {0.f, 0.f, 0.f, 0.f, 0.f, 0.f, 0.f, 0.f};
  f32x8 acc1 = {0.f, 0.f, 0.f, 0.f, 0.f, 0.f, 0.f, 0.f};

  const unsigned bsOff = lds_off32(&Bs[0]);
  const int KT = K >> 6;
  for (int kt = 0; kt < KT; ++kt) {
    const int k0 = kt << 6;
    __syncthreads();
#if HAVE_TDM
    if (wv == 0)   // one DMA per block: weight tile -> LDS (padded rows)
      tdm_load_b_tile(&Wb[(size_t)n0 * ldbK + k0], bsOff, K - k0, N - n0, ldbK);
#else
#pragma unroll
    for (int s = tid; s < 512; s += 256) {
      const int n = s >> 3, seg = (s & 7) << 3;
      *(uint4*)&Bs[n * 72 + seg] =
          *(const uint4*)&Wb[(size_t)(n0 + n) * ldbK + k0 + seg];
    }
#endif
    // ---- stage A tile (overlaps the TDM) ----
    if (AT == 0) {
#pragma unroll
      for (int s = tid; s < 1024; s += 256) {
        const int row = s >> 4;
        const int kk = (s & 15) << 2;
        const float4 v = *(const float4*)&Ab[(size_t)(m0 + row) * lda + (k0 + kk)];
        uint2 p; p.x = pack2bf(v.x, v.y); p.y = pack2bf(v.z, v.w);
        *(uint2*)&As[row * 72 + kk] = p;
      }
    } else {
#pragma unroll
      for (int s = tid; s < 1024; s += 256) {
        const int kk = s >> 4;
        const int mseg = (s & 15) << 2;
        const float4 v = *(const float4*)&Ab[(size_t)(k0 + kk) * lda + (m0 + mseg)];
        As[(mseg + 0) * 72 + kk] = f2bf(v.x);
        As[(mseg + 1) * 72 + kk] = f2bf(v.y);
        As[(mseg + 2) * 72 + kk] = f2bf(v.z);
        As[(mseg + 3) * 72 + kk] = f2bf(v.w);
      }
    }
    if (kt + 1 < KT) {
      const int r = tid >> 2, q = (tid & 3) << 4;
      const float* pa = (AT == 0) ? &Ab[(size_t)(m0 + r) * lda + (k0 + 64 + q)]
                                  : &Ab[(size_t)(k0 + 64 + r) * lda + (m0 + q)];
      __builtin_prefetch(pa, 0, 1);
    }
#if HAVE_TDM
    if (wv == 0) __builtin_amdgcn_s_wait_tensorcnt(0);
#endif
    __syncthreads();
#pragma unroll
    for (int kc = 0; kc < 64; kc += 32) {
      FragU fa, fb0, fb1;
      const int ao = (tr * 16 + lm) * 72 + kc + (hi ? 16 : 0);
      fa.q[0] = *(const uint4*)&As[ao];
      fa.q[1] = *(const uint4*)&As[ao + 8];
      const int kb = kc + (hi ? 16 : 0);
      const int bo0 = (tc * 16 + lm) * 72 + kb;
      fb0.q[0] = *(const uint4*)&Bs[bo0];
      fb0.q[1] = *(const uint4*)&Bs[bo0 + 8];
      const int bo1 = bo0 + 16 * 72;
      fb1.q[0] = *(const uint4*)&Bs[bo1];
      fb1.q[1] = *(const uint4*)&Bs[bo1 + 8];
      acc0 = __builtin_amdgcn_wmma_f32_16x16x32_bf16(false, fa.v, false, fb0.v,
                                                     (short)0, acc0, false, false);
      acc1 = __builtin_amdgcn_wmma_f32_16x16x32_bf16(false, fa.v, false, fb1.v,
                                                     (short)0, acc1, false, false);
    }
  }
  const float* Rb = res ? res + (size_t)blockIdx.z * sC : nullptr;
  float* Cb = Cm + (size_t)blockIdx.z * sC;
#pragma unroll
  for (int t = 0; t < 2; ++t) {
    f32x8 acc;
    if (t) acc = acc1; else acc = acc0;
    const int gn = n0 + (tc + t) * 16 + lm;
#pragma unroll
    for (int r = 0; r < 8; ++r) {
      const int gm = m0 + tr * 16 + r + hi * 8;
      float v = acc[r];
      if (bias) v += bias[gn];
      if (epi == EPI_GELU) v = gelu_exact(v);
      if (epi == EPI_RES)  v += Rb[(size_t)gm * ldc + gn];
      const size_t idx = cT ? ((size_t)gn * ldc + gm) : ((size_t)gm * ldc + gn);
      Cb[idx] = v;
    }
  }
}

// ---------------- generic batched bf16-WMMA GEMM (f32 A and B) ----------------
// BT: B stored (N,K) row-major; BOUND guards M/N/K tails.
template <int BT, bool BOUND>
__global__ __launch_bounds__(256) void gemm_kernel(
    const float* __restrict__ A, int lda, long long sA,
    const float* __restrict__ Bm, int ldb, long long sB,
    const float* __restrict__ bias,
    float* __restrict__ Cm, int ldc, long long sC,
    int M, int N, int K, int epi, float scale) {
  __shared__ __align__(16) unsigned short As[64 * 72];
  __shared__ __align__(16) unsigned short Bs[64 * 72];
  const int tid = threadIdx.x;
  const int wv = tid >> 5, ln = tid & 31;
  const int lm = ln & 15, hi = ln >> 4;
  const int tr = wv & 3;
  const int tc = (wv >> 2) << 1;
  const int m0 = blockIdx.y * 64, n0 = blockIdx.x * 64;
  const float* Ab = A + (size_t)blockIdx.z * sA;
  const float* Bb = Bm + (size_t)blockIdx.z * sB;

  f32x8 acc0 = {0.f, 0.f, 0.f, 0.f, 0.f, 0.f, 0.f, 0.f};
  f32x8 acc1 = {0.f, 0.f, 0.f, 0.f, 0.f, 0.f, 0.f, 0.f};

  const int KT = (K + 63) >> 6;
  for (int kt = 0; kt < KT; ++kt) {
    const int k0 = kt << 6;
    __syncthreads();
#pragma unroll
    for (int s = tid; s < 1024; s += 256) {
      const int row = s >> 4;
      const int kk = (s & 15) << 2;
      const float* src = &Ab[(size_t)(m0 + row) * lda + (k0 + kk)];
      float4 v;
      if (!BOUND) {
        v = *(const float4*)src;
      } else {
        const int gm = m0 + row, gk = k0 + kk;
        if (gm < M && gk + 3 < K) {
          v = *(const float4*)src;
        } else {
          v.x = (gm < M && gk + 0 < K) ? src[0] : 0.f;
          v.y = (gm < M && gk + 1 < K) ? src[1] : 0.f;
          v.z = (gm < M && gk + 2 < K) ? src[2] : 0.f;
          v.w = (gm < M && gk + 3 < K) ? src[3] : 0.f;
        }
      }
      uint2 p; p.x = pack2bf(v.x, v.y); p.y = pack2bf(v.z, v.w);
      *(uint2*)&As[row * 72 + kk] = p;
    }
    if (BT == 1) {
#pragma unroll
      for (int s = tid; s < 1024; s += 256) {
        const int n = s >> 4;
        const int kk = (s & 15) << 2;
        const float* src = &Bb[(size_t)(n0 + n) * ldb + (k0 + kk)];
        float4 v;
        if (!BOUND) {
          v = *(const float4*)src;
        } else {
          const int gn = n0 + n, gk = k0 + kk;
          if (gn < N && gk + 3 < K) {
            v = *(const float4*)src;
          } else {
            v.x = (gn < N && gk + 0 < K) ? src[0] : 0.f;
            v.y = (gn < N && gk + 1 < K) ? src[1] : 0.f;
            v.z = (gn < N && gk + 2 < K) ? src[2] : 0.f;
            v.w = (gn < N && gk + 3 < K) ? src[3] : 0.f;
          }
        }
        uint2 p; p.x = pack2bf(v.x, v.y); p.y = pack2bf(v.z, v.w);
        *(uint2*)&Bs[n * 72 + kk] = p;
      }
    } else {
#pragma unroll
      for (int s = tid; s < 1024; s += 256) {
        const int kk = s >> 4;
        const int nseg = (s & 15) << 2;
        const float* src = &Bb[(size_t)(k0 + kk) * ldb + (n0 + nseg)];
        float4 v;
        if (!BOUND) {
          v = *(const float4*)src;
        } else {
          const int gk = k0 + kk, gn = n0 + nseg;
          if (gk < K && gn + 3 < N) {
            v = *(const float4*)src;
          } else {
            v.x = (gk < K && gn + 0 < N) ? src[0] : 0.f;
            v.y = (gk < K && gn + 1 < N) ? src[1] : 0.f;
            v.z = (gk < K && gn + 2 < N) ? src[2] : 0.f;
            v.w = (gk < K && gn + 3 < N) ? src[3] : 0.f;
          }
        }
        Bs[(nseg + 0) * 72 + kk] = f2bf(v.x);
        Bs[(nseg + 1) * 72 + kk] = f2bf(v.y);
        Bs[(nseg + 2) * 72 + kk] = f2bf(v.z);
        Bs[(nseg + 3) * 72 + kk] = f2bf(v.w);
      }
    }
    __syncthreads();
#pragma unroll
    for (int kc = 0; kc < 64; kc += 32) {
      FragU fa, fb0, fb1;
      const int ao = (tr * 16 + lm) * 72 + kc + (hi ? 16 : 0);
      fa.q[0] = *(const uint4*)&As[ao];
      fa.q[1] = *(const uint4*)&As[ao + 8];
      const int kb = kc + (hi ? 16 : 0);
      const int bo0 = (tc * 16 + lm) * 72 + kb;
      fb0.q[0] = *(const uint4*)&Bs[bo0];
      fb0.q[1] = *(const uint4*)&Bs[bo0 + 8];
      const int bo1 = bo0 + 16 * 72;
      fb1.q[0] = *(const uint4*)&Bs[bo1];
      fb1.q[1] = *(const uint4*)&Bs[bo1 + 8];
      acc0 = __builtin_amdgcn_wmma_f32_16x16x32_bf16(false, fa.v, false, fb0.v,
                                                     (short)0, acc0, false, false);
      acc1 = __builtin_amdgcn_wmma_f32_16x16x32_bf16(false, fa.v, false, fb1.v,
                                                     (short)0, acc1, false, false);
    }
  }
  float* Cb = Cm + (size_t)blockIdx.z * sC;
#pragma unroll
  for (int t = 0; t < 2; ++t) {
    f32x8 acc;
    if (t) acc = acc1; else acc = acc0;
    const int gn = n0 + (tc + t) * 16 + lm;
#pragma unroll
    for (int r = 0; r < 8; ++r) {
      const int gm = m0 + tr * 16 + r + hi * 8;
      if (!BOUND || (gm < M && gn < N)) {
        float v = acc[r];
        if (epi == EPI_SCALE) v *= scale;
        if (bias) v += bias[gn];
        Cb[(size_t)gm * ldc + gn] = v;
      }
    }
  }
}

// ---------------- fused flash cross-attention per (b,h) ----------------
__global__ __launch_bounds__(256) void attn_kernel(const float* __restrict__ qp,
                                                   const float* __restrict__ kp,
                                                   const float* __restrict__ vp,
                                                   float* __restrict__ ctx) {
  __shared__ __align__(16) unsigned short Qs[64 * 72];
  __shared__ __align__(16) unsigned short Ks[64 * 72];
  __shared__ __align__(16) unsigned short Vt[64 * 72];   // [d][l_in_chunk]
  __shared__ __align__(16) unsigned short Ps[64 * 72];   // probs bf16
  __shared__ float Ss[64 * 64];                          // chunk scores f32
  __shared__ float rowmax[64], rowsum[64], rowfac[64];

  const int h = blockIdx.x, b = blockIdx.y;
  const int tid = threadIdx.x, wv = tid >> 5, ln = tid & 31;
  const int lm = ln & 15, hi = ln >> 4;
  const int tr = wv & 3;
  const int tcb = (wv >> 2) << 1;

  const float* qb = qp + ((size_t)b * GGO) * CC + h * 64;
  const float* kb = kp + ((size_t)b * LL) * CC + h * 64;
  const float* vb = vp + ((size_t)b * LL) * CC + h * 64;

#pragma unroll
  for (int s = tid; s < 1024; s += 256) {
    const int m = s >> 4, c = (s & 15) << 2;
    const float4 v = *(const float4*)&qb[(size_t)m * CC + c];
    uint2 p; p.x = pack2bf(v.x, v.y); p.y = pack2bf(v.z, v.w);
    *(uint2*)&Qs[m * 72 + c] = p;
  }
  if (tid < 64) { rowmax[tid] = -3.0e38f; rowsum[tid] = 0.f; }

  f32x8 o0 = {0.f, 0.f, 0.f, 0.f, 0.f, 0.f, 0.f, 0.f};
  f32x8 o1 = {0.f, 0.f, 0.f, 0.f, 0.f, 0.f, 0.f, 0.f};

  for (int cc = 0; cc < 4; ++cc) {
    const int l0 = cc * 64;
    __syncthreads();
#pragma unroll
    for (int s = tid; s < 1024; s += 256) {
      const int l = s >> 4, c = (s & 15) << 2;
      const int gl = l0 + l;
      float4 kv, vv;
      if (gl < LL) {
        kv = *(const float4*)&kb[(size_t)gl * CC + c];
        vv = *(const float4*)&vb[(size_t)gl * CC + c];
      } else {
        kv.x = kv.y = kv.z = kv.w = 0.f;
        vv.x = vv.y = vv.z = vv.w = 0.f;
      }
      uint2 p; p.x = pack2bf(kv.x, kv.y); p.y = pack2bf(kv.z, kv.w);
      *(uint2*)&Ks[l * 72 + c] = p;
      Vt[(c + 0) * 72 + l] = f2bf(vv.x);
      Vt[(c + 1) * 72 + l] = f2bf(vv.y);
      Vt[(c + 2) * 72 + l] = f2bf(vv.z);
      Vt[(c + 3) * 72 + l] = f2bf(vv.w);
    }
    __syncthreads();
#pragma unroll
    for (int t = 0; t < 2; ++t) {
      const int stc = tcb + t;
      f32x8 acc = {0.f, 0.f, 0.f, 0.f, 0.f, 0.f, 0.f, 0.f};
#pragma unroll
      for (int kc = 0; kc < 64; kc += 32) {
        FragU fa, fb;
        const int ao = (tr * 16 + lm) * 72 + kc + (hi ? 16 : 0);
        fa.q[0] = *(const uint4*)&Qs[ao];
        fa.q[1] = *(const uint4*)&Qs[ao + 8];
        const int bo = (stc * 16 + lm) * 72 + kc + (hi ? 16 : 0);
        fb.q[0] = *(const uint4*)&Ks[bo];
        fb.q[1] = *(const uint4*)&Ks[bo + 8];
        acc = __builtin_amdgcn_wmma_f32_16x16x32_bf16(false, fa.v, false, fb.v,
                                                      (short)0, acc, false, false);
      }
      const int gn = stc * 16 + lm;
#pragma unroll
      for (int r = 0; r < 8; ++r) {
        const int gm = tr * 16 + r + hi * 8;
        float v = acc[r] * 0.125f;
        if (l0 + gn >= LL) v = -3.0e38f;
        Ss[gm * 64 + gn] = v;
      }
    }
    __syncthreads();
    if (tid < 64) {
      const float om = rowmax[tid];
      float cm = -3.0e38f;
      for (int j = 0; j < 64; ++j) cm = fmaxf(cm, Ss[tid * 64 + j]);
      const float nm = fmaxf(om, cm);
      const float fac = __expf(om - nm);
      float csum = 0.f;
      for (int j = 0; j < 64; ++j) {
        const float p = __expf(Ss[tid * 64 + j] - nm);
        csum += p;
        Ps[tid * 72 + j] = f2bf(p);
      }
      rowmax[tid] = nm;
      rowsum[tid] = rowsum[tid] * fac + csum;
      rowfac[tid] = fac;
    }
    __syncthreads();
#pragma unroll
    for (int r = 0; r < 8; ++r) {
      const float f = rowfac[tr * 16 + r + hi * 8];
      o0[r] *= f; o1[r] *= f;
    }
#pragma unroll
    for (int kc = 0; kc < 64; kc += 32) {
      FragU fa, fb0, fb1;
      const int ao = (tr * 16 + lm) * 72 + kc + (hi ? 16 : 0);
      fa.q[0] = *(const uint4*)&Ps[ao];
      fa.q[1] = *(const uint4*)&Ps[ao + 8];
      const int kb2 = kc + (hi ? 16 : 0);
      const int bo0 = (tcb * 16 + lm) * 72 + kb2;
      fb0.q[0] = *(const uint4*)&Vt[bo0];
      fb0.q[1] = *(const uint4*)&Vt[bo0 + 8];
      const int bo1 = bo0 + 16 * 72;
      fb1.q[0] = *(const uint4*)&Vt[bo1];
      fb1.q[1] = *(const uint4*)&Vt[bo1 + 8];
      o0 = __builtin_amdgcn_wmma_f32_16x16x32_bf16(false, fa.v, false, fb0.v,
                                                   (short)0, o0, false, false);
      o1 = __builtin_amdgcn_wmma_f32_16x16x32_bf16(false, fa.v, false, fb1.v,
                                                   (short)0, o1, false, false);
    }
  }
  float* cb = ctx + ((size_t)b * GGO) * CC + h * 64;
#pragma unroll
  for (int t = 0; t < 2; ++t) {
    f32x8 acc;
    if (t) acc = o1; else acc = o0;
    const int gn = (tcb + t) * 16 + lm;
#pragma unroll
    for (int r = 0; r < 8; ++r) {
      const int gm = tr * 16 + r + hi * 8;
      cb[(size_t)gm * CC + gn] = acc[r] / rowsum[gm];
    }
  }
}

// ------------- assign softmax over groups + straight-through one-hot -------------
__global__ __launch_bounds__(256) void assign_softmax_kernel(
    const float* __restrict__ raw, float* __restrict__ soft,
    float* __restrict__ attn) {
  const int b = blockIdx.x, l = threadIdx.x;
  if (l >= LL) return;
  const float* rb = raw + (size_t)b * GGO * LL;
  float mx = -3.0e38f; int am = 0;
  for (int g = 0; g < GGO; ++g) {
    const float v = rb[g * LL + l];
    if (v > mx) { mx = v; am = g; }
  }
  float sum = 0.f;
  for (int g = 0; g < GGO; ++g) sum += __expf(rb[g * LL + l] - mx);
  const float inv = 1.f / sum;
  float* sb = soft + (size_t)b * GGO * LL;
  float* ab = attn + (size_t)b * GGO * LL;
  for (int g = 0; g < GGO; ++g) {
    const float s = __expf(rb[g * LL + l] - mx) * inv;
    sb[g * LL + l] = s;
    const float oh = (g == am) ? 1.f : 0.f;
    ab[g * LL + l] = (oh - s) + s;
  }
}

// ------------- normalize each (b,g) row by (sum_l + 1) -------------
__global__ __launch_bounds__(256) void rownorm_kernel(float* __restrict__ attn) {
  __shared__ float red[256];
  const int bg = blockIdx.x;
  float* row = attn + (size_t)bg * LL;
  float s = 0.f;
  for (int i = threadIdx.x; i < LL; i += 256) s += row[i];
  red[threadIdx.x] = s;
  __syncthreads();
  for (int st = 128; st > 0; st >>= 1) {
    if (threadIdx.x < st) red[threadIdx.x] += red[threadIdx.x + st];
    __syncthreads();
  }
  const float inv = 1.f / (red[0] + 1.0f);   // ASSIGN_EPS = 1.0
  for (int i = threadIdx.x; i < LL; i += 256) row[i] *= inv;
}

// ================= host orchestration =================
template <int AT>
static void launch_wgemm(hipStream_t stream,
                         const float* A, int lda, long long sA,
                         const unsigned short* Wb, int ldbK,
                         const float* bias, const float* res,
                         float* Cm, int ldc, long long sC, int cT,
                         int M, int N, int K, int batch, int epi) {
  dim3 grid((N + 63) / 64, (M + 63) / 64, batch);
  hipLaunchKernelGGL((gemm_wb_kernel<AT>), grid, dim3(256), 0, stream,
                     A, lda, sA, Wb, ldbK, bias, res, Cm, ldc, sC, cT,
                     M, N, K, epi);
}

extern "C" void kernel_launch(void* const* d_in, const int* in_sizes, int n_in,
                              void* d_out, int out_size, void* d_ws, size_t ws_size,
                              hipStream_t stream) {
  (void)in_sizes; (void)n_in; (void)out_size; (void)ws_size;
  const float* img  = (const float*)d_in[0];
  const float* grp  = (const float*)d_in[1];
  const float* ntg  = (const float*)d_in[2];
  const float* ntb  = (const float*)d_in[3];
  const float* nxg  = (const float*)d_in[4];
  const float* nxb  = (const float*)d_in[5];
  const float* mi1w = (const float*)d_in[6];
  const float* mi1b = (const float*)d_in[7];
  const float* mi2w = (const float*)d_in[8];
  const float* mi2b = (const float*)d_in[9];
  const float* nptg = (const float*)d_in[10];
  const float* nptb = (const float*)d_in[11];
  const float* aqw  = (const float*)d_in[12];
  const float* aqb  = (const float*)d_in[13];
  const float* akw  = (const float*)d_in[14];
  const float* akb  = (const float*)d_in[15];
  const float* avw  = (const float*)d_in[16];
  const float* avb  = (const float*)d_in[17];
  const float* aow  = (const float*)d_in[18];
  const float* aob  = (const float*)d_in[19];
  const float* n2g  = (const float*)d_in[20];
  const float* n2b  = (const float*)d_in[21];
  const float* pf1w = (const float*)d_in[22];
  const float* pf1b = (const float*)d_in[23];
  const float* pf2w = (const float*)d_in[24];
  const float* pf2b = (const float*)d_in[25];
  const float* npg  = (const float*)d_in[26];
  const float* npb  = (const float*)d_in[27];
  const float* sqw  = (const float*)d_in[28];
  const float* sqb  = (const float*)d_in[29];
  const float* skw  = (const float*)d_in[30];
  const float* skb  = (const float*)d_in[31];
  const float* svw  = (const float*)d_in[32];
  const float* svb  = (const float*)d_in[33];
  const float* spw  = (const float*)d_in[34];
  const float* spb  = (const float*)d_in[35];
  const float* nnxg = (const float*)d_in[36];
  const float* nnxb = (const float*)d_in[37];
  const float* mc1w = (const float*)d_in[38];
  const float* mc1b = (const float*)d_in[39];
  const float* mc2w = (const float*)d_in[40];
  const float* mc2b = (const float*)d_in[41];

  float* out = (float*)d_out;
  const long long NX = (long long)BB * GGO * CC;
  float* soft_out = out + NX;

  // ---- workspace arena ----
  const long long BLC = (long long)BB * LL * CC;
  const long long BGC = (long long)BB * GGO * CC;
  const long long BGF = (long long)BB * GGO * FFD;
  const long long BGL = (long long)BB * GGO * LL;
  float* w = (float*)d_ws;
  size_t off = 0;
  auto take = [&](long long n) { float* p = w + off; off += (size_t)n; return p; };
  auto takeh = [&](long long n) {
    unsigned short* p = (unsigned short*)(w + off);
    off += (size_t)((n + 1) / 2);
    return p;
  };
  float* x_ln = take(BLC);
  float* big1 = take(BLC);
  float* big2 = take(BLC);
  float* g0   = take(BGC);
  float* g2   = take(BGC);
  float* g3   = take(BGC);
  float* g4   = take(BGC);
  float* g6   = take(BGC);
  float* hb   = take(BGF);
  float* r0   = take(BGL);
  float* r1   = take(BGL);
  // bf16 transposed weights (N,K)
  unsigned short* mi1t = takeh((long long)TOKD * GG);
  unsigned short* mi2t = takeh((long long)GGO * TOKD);
  unsigned short* aqt  = takeh((long long)CC * CC);
  unsigned short* akt  = takeh((long long)CC * CC);
  unsigned short* avt  = takeh((long long)CC * CC);
  unsigned short* aot  = takeh((long long)CC * CC);
  unsigned short* sqt  = takeh((long long)CC * CC);
  unsigned short* skt  = takeh((long long)CC * CC);
  unsigned short* svt  = takeh((long long)CC * CC);
  unsigned short* spt  = takeh((long long)CC * CC);
  unsigned short* pf1t = takeh((long long)FFD * CC);
  unsigned short* pf2t = takeh((long long)CC * FFD);
  unsigned short* mc1t = takeh((long long)CHD * CC);
  unsigned short* mc2t = takeh((long long)CC * CHD);

  dim3 blk(256);
  const int MG = BB * GGO;    // 16384
  const int ML = BB * LL;     // 50176

  // ---- weight prep: transpose + bf16 ----
  auto wprep = [&](const float* in, unsigned short* o, int K, int N) {
    hipLaunchKernelGGL(wprep_kernel, dim3((N * K + 255) / 256), blk, 0, stream,
                       in, o, K, N);
  };
  wprep(mi1w, mi1t, GG, TOKD);
  wprep(mi2w, mi2t, TOKD, GGO);
  wprep(aqw, aqt, CC, CC);  wprep(akw, akt, CC, CC);
  wprep(avw, avt, CC, CC);  wprep(aow, aot, CC, CC);
  wprep(sqw, sqt, CC, CC);  wprep(skw, skt, CC, CC);
  wprep(svw, svt, CC, CC);  wprep(spw, spt, CC, CC);
  wprep(pf1w, pf1t, CC, FFD);
  wprep(pf2w, pf2t, FFD, CC);
  wprep(mc1w, mc1t, CC, CHD);
  wprep(mc2w, mc2t, FFD, CC);

  // LN of inputs
  hipLaunchKernelGGL(ln_kernel, dim3(BB * GG), blk, 0, stream, grp, ntg, ntb, g0, CC);
  hipLaunchKernelGGL(ln_kernel, dim3(ML),      blk, 0, stream, img, nxg, nxb, x_ln, CC);
  // mixer MLP over token axis
  launch_wgemm<1>(stream, g0, CC, (long long)GG * CC, mi1t, GG, mi1b, nullptr,
                  big1, TOKD, (long long)CC * TOKD, 0, CC, TOKD, GG, BB, EPI_GELU);
  launch_wgemm<0>(stream, big1, TOKD, (long long)CC * TOKD, mi2t, TOKD, mi2b, nullptr,
                  g2, CC, (long long)GGO * CC, 1, CC, GGO, TOKD, BB, EPI_NONE);
  hipLaunchKernelGGL(ln_kernel, dim3(MG), blk, 0, stream, g2, nptg, nptb, g3, CC);
  // cross-attention projections
  launch_wgemm<0>(stream, g3, CC, 0, aqt, CC, aqb, nullptr, g4, CC, 0, 0,
                  MG, CC, CC, 1, EPI_NONE);
  launch_wgemm<0>(stream, x_ln, CC, 0, akt, CC, akb, nullptr, big1, CC, 0, 0,
                  ML, CC, CC, 1, EPI_NONE);
  launch_wgemm<0>(stream, x_ln, CC, 0, avt, CC, avb, nullptr, big2, CC, 0, 0,
                  ML, CC, CC, 1, EPI_NONE);
  hipLaunchKernelGGL(attn_kernel, dim3(HH, BB), blk, 0, stream, g4, big1, big2, g6);
  // output projection + residual
  launch_wgemm<0>(stream, g6, CC, 0, aot, CC, aob, g3, g2, CC, 0, 0,
                  MG, CC, CC, 1, EPI_RES);
  // MLP block
  hipLaunchKernelGGL(ln_kernel, dim3(MG), blk, 0, stream, g2, n2g, n2b, g4, CC);
  launch_wgemm<0>(stream, g4, CC, 0, pf1t, CC, pf1b, nullptr, hb, FFD, 0, 0,
                  MG, FFD, CC, 1, EPI_GELU);
  launch_wgemm<0>(stream, hb, FFD, 0, pf2t, FFD, pf2b, g2, g6, CC, 0, 0,
                  MG, CC, FFD, 1, EPI_RES);
  hipLaunchKernelGGL(ln_kernel, dim3(MG), blk, 0, stream, g6, npg, npb, g3, CC);
  // assign projections
  launch_wgemm<0>(stream, g3, CC, 0, sqt, CC, sqb, nullptr, g4, CC, 0, 0,
                  MG, CC, CC, 1, EPI_NONE);
  launch_wgemm<0>(stream, x_ln, CC, 0, skt, CC, skb, nullptr, big1, CC, 0, 0,
                  ML, CC, CC, 1, EPI_NONE);
  launch_wgemm<0>(stream, x_ln, CC, 0, svt, CC, svb, nullptr, big2, CC, 0, 0,
                  ML, CC, CC, 1, EPI_NONE);
  // raw = qq @ kk^T * C^-0.5  (batched over B; activation B, f32)
  {
    dim3 grid((LL + 63) / 64, (GGO + 63) / 64, BB);
    hipLaunchKernelGGL((gemm_kernel<1, true>), grid, blk, 0, stream,
                       g4, CC, (long long)GGO * CC, big1, CC, (long long)LL * CC,
                       (const float*)nullptr, r0, LL, (long long)GGO * LL,
                       GGO, LL, CC, EPI_SCALE, 0.051031036307982884f);
  }
  hipLaunchKernelGGL(assign_softmax_kernel, dim3(BB), blk, 0, stream, r0, soft_out, r1);
  hipLaunchKernelGGL(rownorm_kernel, dim3(BB * GGO), blk, 0, stream, r1);
  // out = attn @ vv  (batched over B; activation B, f32)
  {
    dim3 grid((CC + 63) / 64, (GGO + 63) / 64, BB);
    hipLaunchKernelGGL((gemm_kernel<0, true>), grid, blk, 0, stream,
                       r1, LL, (long long)GGO * LL, big2, CC, (long long)LL * CC,
                       (const float*)nullptr, g2, CC, (long long)GGO * CC,
                       GGO, CC, LL, EPI_NONE, 1.f);
  }
  // sp projection + residual gt4
  launch_wgemm<0>(stream, g2, CC, 0, spt, CC, spb, g3, g4, CC, 0, 0,
                  MG, CC, CC, 1, EPI_RES);
  // channel MLP
  hipLaunchKernelGGL(ln_kernel, dim3(MG), blk, 0, stream, g4, nnxg, nnxb, g3, CC);
  launch_wgemm<0>(stream, g3, CC, 0, mc1t, CC, mc1b, nullptr, hb, CHD, 0, 0,
                  MG, CHD, CC, 1, EPI_GELU);
  launch_wgemm<0>(stream, hb, CHD, 0, mc2t, CHD, mc2b, g4, out, CC, 0, 0,
                  MG, CC, CHD, 1, EPI_RES);
}
